// HymbaSdpaAttention_40561671144013
// MI455X (gfx1250) — matrix operand order repaired and checked by reading, not compiled, not run on hardware
//
#include <hip/hip_runtime.h>
#include <hip/hip_bf16.h>

// ---------------------------------------------------------------------------
// Hymba GQA attention for MI455X (gfx1250): bf16 WMMA everywhere.
// B=2, S=2048, D=2048, NH=16, NKV=4, HD=128.
// ---------------------------------------------------------------------------

typedef __bf16 bf16_t;
typedef __attribute__((ext_vector_type(16))) __bf16 v16bf;
typedef __attribute__((ext_vector_type(8)))  float  v8f;

#define BB   2
#define SS   2048
#define DD   2048
#define NH   16
#define NKV  4
#define HD   128
#define MROWS (BB * SS)          // 4096
#define NEGINF (-1e30f)

// A-fragment (16x32, 16-bit) k index for element e, lane-half h (ISA 7.12.2)
__device__ __forceinline__ int kofA(int e, int h) {
    return (e < 8) ? (h * 8 + e) : (16 + h * 8 + (e - 8));
}
// B-fragment (32x16, 16-bit) k index: same order as A-fragment
__device__ __forceinline__ int kofB(int e, int h) { return (e < 8) ? (h * 8 + e) : (16 + h * 8 + (e - 8)); }

__device__ __forceinline__ v8f wmma_bf16(v16bf a, v16bf b, v8f c) {
    return __builtin_amdgcn_wmma_f32_16x16x32_bf16(
        false, a, false, b, (short)0, c, false, false);
}

// ---------------------------------------------------------------------------
// Pack fp32 row-major [M x K] into A-fragment bf16 tiles [mt][kt][lane][16]
// ---------------------------------------------------------------------------
__global__ void pack_a_kernel(const float* __restrict__ src, bf16_t* __restrict__ dst,
                              int Ktiles, int ld, int total) {
    int idx = blockIdx.x * blockDim.x + threadIdx.x;
    if (idx >= total) return;
    int lane = idx & 31;
    int tile = idx >> 5;
    int kt   = tile % Ktiles;
    int mt   = tile / Ktiles;
    int m    = mt * 16 + (lane & 15);
    int half = lane >> 4;
    const float* row = src + (size_t)m * ld + kt * 32;
    v16bf d;
#pragma unroll
    for (int e = 0; e < 16; ++e) d[e] = (bf16_t)row[kofA(e, half)];
    *(v16bf*)(dst + (size_t)idx * 16) = d;
}

// Pack fp32 row-major [K x N] into B-fragment bf16 tiles [nt][kt][lane][16]
__global__ void pack_b_kernel(const float* __restrict__ src, bf16_t* __restrict__ dst,
                              int Ktiles, int ld, int total) {
    int idx = blockIdx.x * blockDim.x + threadIdx.x;
    if (idx >= total) return;
    int lane = idx & 31;
    int tile = idx >> 5;
    int kt   = tile % Ktiles;
    int nt   = tile / Ktiles;
    int n    = nt * 16 + (lane & 15);
    int half = lane >> 4;
    v16bf d;
#pragma unroll
    for (int e = 0; e < 16; ++e) {
        int k = kt * 32 + kofB(e, half);
        d[e] = (bf16_t)src[(size_t)k * ld + n];
    }
    *(v16bf*)(dst + (size_t)idx * 16) = d;
}

// ---------------------------------------------------------------------------
// GEMM on packed fragments: each wave = 64(m) x 32(n), block = 8 waves = 128x128
// C is fp32 row-major with leading dim ldc.
// ---------------------------------------------------------------------------
__global__ void __launch_bounds__(256)
gemm_packed_kernel(const bf16_t* __restrict__ Ap, const bf16_t* __restrict__ Bp,
                   float* __restrict__ C, int Ktiles, int ldc) {
    int lane = threadIdx.x & 31;
    int w    = threadIdx.x >> 5;
    int wm   = w >> 2, wn = w & 3;
    int m0t  = blockIdx.y * 8 + wm * 4;
    int n0t  = blockIdx.x * 8 + wn * 2;
    int half = lane >> 4, nl = lane & 15;

    size_t tStride = (size_t)Ktiles * 512;   // elems per tile-row/col step
    const bf16_t* aptr = Ap + (size_t)m0t * tStride + lane * 16;
    const bf16_t* bptr = Bp + (size_t)n0t * tStride + lane * 16;

    v8f acc[4][2] = {};
    for (int kt = 0; kt < Ktiles; ++kt) {
        v16bf a[4], bf[2];
#pragma unroll
        for (int i = 0; i < 4; ++i)
            a[i] = *(const v16bf*)(aptr + (size_t)i * tStride + (size_t)kt * 512);
#pragma unroll
        for (int j = 0; j < 2; ++j)
            bf[j] = *(const v16bf*)(bptr + (size_t)j * tStride + (size_t)kt * 512);
#pragma unroll
        for (int i = 0; i < 4; ++i)
#pragma unroll
            for (int j = 0; j < 2; ++j)
                acc[i][j] = wmma_bf16(a[i], bf[j], acc[i][j]);
    }
#pragma unroll
    for (int i = 0; i < 4; ++i)
#pragma unroll
        for (int j = 0; j < 2; ++j)
#pragma unroll
            for (int r = 0; r < 8; ++r) {
                int row = (m0t + i) * 16 + r + 8 * half;
                int col = (n0t + j) * 16 + nl;
                C[(size_t)row * ldc + col] = acc[i][j][r];
            }
}

// ---------------------------------------------------------------------------
// RoPE cos/sin table: tab[b*S+s][i] = {cos, sin}(pos * 10000^(-i/64)), i<64
// ---------------------------------------------------------------------------
__global__ void rope_table_kernel(const int* __restrict__ pos_ids, float2* __restrict__ tab) {
    int idx = blockIdx.x * blockDim.x + threadIdx.x;
    if (idx >= BB * SS * 64) return;
    int i   = idx & 63;
    int row = idx >> 6;
    float pos  = (float)pos_ids[row];
    float invf = powf(10000.0f, -(float)i / 64.0f);
    float ang  = pos * invf;
    tab[idx] = make_float2(cosf(ang), sinf(ang));
}

// ---------------------------------------------------------------------------
// RoPE + pack Q as A-fragments per (b,h): tiles [(b,h)][s/16][hd/32]
// qkv scratch layout: row b*S+s, cols: q at h*128, k at 2048+kvh*128, v at 2560+kvh*128
// ---------------------------------------------------------------------------
__global__ void rope_pack_q_kernel(const float* __restrict__ qkv, const float2* __restrict__ tab,
                                   bf16_t* __restrict__ Qp) {
    int idx = blockIdx.x * blockDim.x + threadIdx.x;  // < B*NH*128*4*32
    if (idx >= BB * NH * 128 * 4 * 32) return;
    int lane = idx & 31;
    int tile = idx >> 5;                 // ((b*16+h)*128 + mt)*4 + kt
    int kt = tile & 3;
    int mt = (tile >> 2) & 127;
    int bh = tile >> 9;
    int h = bh & 15, b = bh >> 4;
    int s = mt * 16 + (lane & 15);
    int half = lane >> 4;
    int row = b * SS + s;
    const float*  src  = qkv + (size_t)row * 3072 + h * HD;
    const float2* trow = tab + (size_t)row * 64;
    v16bf d;
#pragma unroll
    for (int e = 0; e < 16; ++e) {
        int hd = kt * 32 + kofA(e, half);
        float2 cs = trow[hd & 63];
        float v = src[hd];
        float o = (hd < 64) ? (v * cs.x - src[hd + 64] * cs.y)
                            : (v * cs.x + src[hd - 64] * cs.y);
        d[e] = (bf16_t)o;
    }
    *(v16bf*)(Qp + (size_t)idx * 16) = d;
}

// RoPE + pack K^T as B-fragments per (b,kvh): tiles [(b,kvh)][key/16][hd/32]
__global__ void rope_pack_k_kernel(const float* __restrict__ qkv, const float2* __restrict__ tab,
                                   bf16_t* __restrict__ Kp) {
    int idx = blockIdx.x * blockDim.x + threadIdx.x;  // < B*NKV*128*4*32
    if (idx >= BB * NKV * 128 * 4 * 32) return;
    int lane = idx & 31;
    int tile = idx >> 5;                 // ((b*4+kvh)*128 + nt)*4 + kt
    int kt = tile & 3;
    int nt = (tile >> 2) & 127;
    int bkv = tile >> 9;
    int kvh = bkv & 3, b = bkv >> 2;
    int s = nt * 16 + (lane & 15);       // key position
    int half = lane >> 4;
    int row = b * SS + s;
    const float*  src  = qkv + (size_t)row * 3072 + 2048 + kvh * HD;
    const float2* trow = tab + (size_t)row * 64;
    v16bf d;
#pragma unroll
    for (int e = 0; e < 16; ++e) {
        int hd = kt * 32 + kofB(e, half);
        float2 cs = trow[hd & 63];
        float v = src[hd];
        float o = (hd < 64) ? (v * cs.x - src[hd + 64] * cs.y)
                            : (v * cs.x + src[hd - 64] * cs.y);
        d[e] = (bf16_t)o;
    }
    *(v16bf*)(Kp + (size_t)idx * 16) = d;
}

// Pack V as B-fragments per (b,kvh): tiles [(b,kvh)][hd/16][key/32]
__global__ void pack_v_kernel(const float* __restrict__ qkv, bf16_t* __restrict__ Vp) {
    int idx = blockIdx.x * blockDim.x + threadIdx.x;  // < B*NKV*8*64*32
    if (idx >= BB * NKV * 8 * 64 * 32) return;
    int lane = idx & 31;
    int tile = idx >> 5;                 // ((b*4+kvh)*8 + nt)*64 + kt
    int kt = tile & 63;
    int nt = (tile >> 6) & 7;
    int bkv = tile >> 9;
    int kvh = bkv & 3, b = bkv >> 2;
    int hd = nt * 16 + (lane & 15);
    int half = lane >> 4;
    v16bf d;
#pragma unroll
    for (int e = 0; e < 16; ++e) {
        int key = kt * 32 + kofB(e, half);
        d[e] = (bf16_t)qkv[(size_t)(b * SS + key) * 3072 + 2560 + kvh * HD + hd];
    }
    *(v16bf*)(Vp + (size_t)idx * 16) = d;
}

// ---------------------------------------------------------------------------
// Flash attention: one wave per (b, h, 16-query tile). Online softmax in f32.
// Output written directly as A-fragments for the final GEMM (M=4096, K=2048).
// ---------------------------------------------------------------------------
__global__ void __launch_bounds__(128)
flash_kernel(const bf16_t* __restrict__ Qp, const bf16_t* __restrict__ Kp,
             const bf16_t* __restrict__ Vp, bf16_t* __restrict__ Op) {
    __shared__ float lds[4][16 * 128];   // per-wave staging (P transpose / O pack)
    int lane = threadIdx.x & 31;
    int w    = threadIdx.x >> 5;
    int task = blockIdx.x * 4 + w;       // < B*NH*128 = 4096
    int qt  = task & 127;
    int h   = (task >> 7) & 15;
    int b   = task >> 11;
    int kvh = h >> 2;
    int half = lane >> 4, nl = lane & 15;
    float* sm = lds[w];

    // Q fragments: 16 rows x 128 hd, resident for the whole key loop
    const bf16_t* qbase = Qp + (size_t)(((b * 16 + h) * 128 + qt) * 4) * 512 + lane * 16;
    v16bf qf[4];
#pragma unroll
    for (int kk = 0; kk < 4; ++kk) qf[kk] = *(const v16bf*)(qbase + (size_t)kk * 512);

    const bf16_t* kbase = Kp + (size_t)((b * 4 + kvh) * 512) * 512 + lane * 16;
    const bf16_t* vbase = Vp + (size_t)((b * 4 + kvh) * 512) * 512 + lane * 16;

    v8f o[8] = {};
    float mi[8], li[8];
#pragma unroll
    for (int r = 0; r < 8; ++r) { mi[r] = NEGINF; li[r] = 0.0f; }

    const float scale = 0.08838834764831845f;  // 1/sqrt(128)
    int qend  = qt * 16 + 15;
    int steps = qend / 32 + 1;

    for (int st = 0; st < steps; ++st) {
        int nt0 = st * 2, nt1 = st * 2 + 1;
        bool use1 = (nt1 * 16 <= qend);
        v8f s0 = {}, s1 = {};
#pragma unroll
        for (int kk = 0; kk < 4; ++kk) {
            v16bf kb0 = *(const v16bf*)(kbase + ((size_t)nt0 * 4 + kk) * 512);
            s0 = wmma_bf16(qf[kk], kb0, s0);
        }
        if (use1) {
#pragma unroll
            for (int kk = 0; kk < 4; ++kk) {
                v16bf kb1 = *(const v16bf*)(kbase + ((size_t)nt1 * 4 + kk) * 512);
                s1 = wmma_bf16(qf[kk], kb1, s1);
            }
        }
        int key0 = nt0 * 16 + nl, key1 = nt1 * 16 + nl;
#pragma unroll
        for (int r = 0; r < 8; ++r) {
            int qrow = qt * 16 + r + 8 * half;
            float v0 = (key0 <= qrow) ? s0[r] * scale : NEGINF;
            float v1 = (use1 && key1 <= qrow) ? s1[r] * scale : NEGINF;
            float mx = fmaxf(fmaxf(v0, v1), mi[r]);
#pragma unroll
            for (int off = 8; off >= 1; off >>= 1)
                mx = fmaxf(mx, __shfl_xor(mx, off, 32));   // stays within 16-lane half
            float alpha = __expf(mi[r] - mx);
            mi[r] = mx;
            float e0 = __expf(v0 - mx);
            float e1 = __expf(v1 - mx);
            float rs = e0 + e1;
#pragma unroll
            for (int off = 8; off >= 1; off >>= 1)
                rs += __shfl_xor(rs, off, 32);
            li[r] = li[r] * alpha + rs;
#pragma unroll
            for (int nt = 0; nt < 8; ++nt) o[nt][r] = o[nt][r] * alpha;
            int mrow = r + 8 * half;
            sm[mrow * 32 + nl]      = e0;   // P in [m][k] order, k in [0,32)
            sm[mrow * 32 + 16 + nl] = e1;
        }
        asm volatile("s_wait_dscnt 0" ::: "memory");
        // Re-read P as an A-fragment (16x32) in bf16
        v16bf pa;
#pragma unroll
        for (int e = 0; e < 16; ++e) pa[e] = (bf16_t)sm[nl * 32 + kofA(e, half)];
        // P @ V over 8 hd-tiles
#pragma unroll
        for (int nt = 0; nt < 8; ++nt) {
            v16bf vb = *(const v16bf*)(vbase + ((size_t)nt * 64 + st) * 512);
            o[nt] = wmma_bf16(pa, vb, o[nt]);
        }
    }

    // Normalize and stage 16x128 O tile, then emit as A-fragments for GEMM2.
#pragma unroll
    for (int nt = 0; nt < 8; ++nt)
#pragma unroll
        for (int r = 0; r < 8; ++r) {
            int mrow = r + 8 * half;
            sm[mrow * 128 + nt * 16 + nl] = o[nt][r] * (1.0f / li[r]);
        }
    asm volatile("s_wait_dscnt 0" ::: "memory");
    int mtileG = b * 128 + qt;                   // global row tile of (b,s)
    bf16_t* obase = Op + (((size_t)mtileG * 64 + h * 4) * 32 + lane) * 16;
#pragma unroll
    for (int kk = 0; kk < 4; ++kk) {
        v16bf d;
#pragma unroll
        for (int e = 0; e < 16; ++e)
            d[e] = (bf16_t)sm[nl * 128 + kk * 32 + kofA(e, half)];
        *(v16bf*)(obase + (size_t)kk * 512) = d;
    }
}

// ---------------------------------------------------------------------------
extern "C" void kernel_launch(void* const* d_in, const int* in_sizes, int n_in,
                              void* d_out, int out_size, void* d_ws, size_t ws_size,
                              hipStream_t stream) {
    (void)in_sizes; (void)n_in; (void)out_size; (void)ws_size;
    const float* x   = (const float*)d_in[0];
    const float* wq  = (const float*)d_in[1];
    const float* wk  = (const float*)d_in[2];
    const float* wv  = (const float*)d_in[3];
    const float* wo  = (const float*)d_in[4];
    const int*   pos = (const int*)  d_in[5];
    float* out = (float*)d_out;

    char* ws = (char*)d_ws;
    size_t off = 0;
    auto take = [&](size_t bytes) -> char* {
        char* p = ws + off;
        off = (off + bytes + 255) & ~(size_t)255;
        return p;
    };
    bf16_t* Xa   = (bf16_t*)take((size_t)256 * 64 * 512 * 2);  // X A-frags
    bf16_t* Wqkv = (bf16_t*)take((size_t)192 * 64 * 512 * 2);  // [Wq|Wk|Wv] B-frags
    bf16_t* Wo   = (bf16_t*)take((size_t)128 * 64 * 512 * 2);  // Wo B-frags
    float*  QKV  = (float*) take((size_t)MROWS * 3072 * 4);    // fused QKV f32
    float2* Tab  = (float2*)take((size_t)MROWS * 64 * 8);      // rope table
    bf16_t* Qp   = (bf16_t*)take((size_t)32 * 512 * 512 * 2);  // Q A-frags (b,h)
    bf16_t* Kp   = (bf16_t*)take((size_t)8  * 512 * 512 * 2);  // K^T B-frags
    bf16_t* Vp   = (bf16_t*)take((size_t)8  * 512 * 512 * 2);  // V B-frags
    bf16_t* Op   = (bf16_t*)take((size_t)256 * 64 * 512 * 2);  // O A-frags

    // 1) pack X and weights to fragment layouts (bf16)
    pack_a_kernel<<<(256 * 64 * 32) / 256, 256, 0, stream>>>(x, Xa, 64, DD, 256 * 64 * 32);
    pack_b_kernel<<<(128 * 64 * 32) / 256, 256, 0, stream>>>(wq, Wqkv, 64, 2048, 128 * 64 * 32);
    pack_b_kernel<<<(32  * 64 * 32) / 256, 256, 0, stream>>>(wk, Wqkv + (size_t)128 * 64 * 512, 64, 512, 32 * 64 * 32);
    pack_b_kernel<<<(32  * 64 * 32) / 256, 256, 0, stream>>>(wv, Wqkv + (size_t)160 * 64 * 512, 64, 512, 32 * 64 * 32);
    pack_b_kernel<<<(128 * 64 * 32) / 256, 256, 0, stream>>>(wo, Wo, 64, 2048, 128 * 64 * 32);

    // 2) fused QKV projection: [4096 x 2048] x [2048 x 3072]
    {
        dim3 g(3072 / 128, 4096 / 128);
        gemm_packed_kernel<<<g, 256, 0, stream>>>(Xa, Wqkv, QKV, 64, 3072);
    }

    // 3) RoPE table + rotate/repack Q, K; repack V
    rope_table_kernel<<<(MROWS * 64) / 256, 256, 0, stream>>>(pos, Tab);
    rope_pack_q_kernel<<<(BB * NH * 128 * 4 * 32) / 256, 256, 0, stream>>>(QKV, Tab, Qp);
    rope_pack_k_kernel<<<(BB * NKV * 128 * 4 * 32) / 256, 256, 0, stream>>>(QKV, Tab, Kp);
    pack_v_kernel<<<(BB * NKV * 8 * 64 * 32) / 256, 256, 0, stream>>>(QKV, Vp);

    // 4) causal flash attention (one wave per (b,h,q-tile))
    flash_kernel<<<(BB * NH * 128) / 4, 128, 0, stream>>>(Qp, Kp, Vp, Op);

    // 5) output projection: [4096 x 2048] x [2048 x 2048] -> d_out
    {
        dim3 g(2048 / 128, 4096 / 128);
        gemm_packed_kernel<<<g, 256, 0, stream>>>(Op, Wo, out, 64, 2048);
    }
}
